// Block_44358422233377
// MI455X (gfx1250) — matrix operand order repaired and checked by reading, not compiled, not run on hardware
//
#include <hip/hip_runtime.h>
#include <hip/hip_bf16.h>

// ---------------------------------------------------------------------------
// Transformer block forward for MI455X (gfx1250, wave32, WMMA).
// B=4, N=1024, C=1024, H=16, HD=64, FF=4096.  All GEMMs routed through
// v_wmma_f32_16x16x32_bf16 (fp32 accumulate). Flash-style attention with
// v_permlane16_b32 softmax reductions. GEMM tiles staged into LDS by the
// Tensor Data Mover (tensor_load_to_lds + s_wait_tensorcnt), double-buffered.
// ---------------------------------------------------------------------------

typedef __attribute__((ext_vector_type(16))) __bf16 bf16x16;
typedef __attribute__((ext_vector_type(8)))  float  fx8;
typedef unsigned int __attribute__((ext_vector_type(4))) uint4v;
typedef int          __attribute__((ext_vector_type(4))) int4v;
typedef int          __attribute__((ext_vector_type(8))) int8v;

#if defined(__has_builtin)
#  if __has_builtin(__builtin_amdgcn_tensor_load_to_lds)
#    define USE_TDM 1
#  else
#    define USE_TDM 0
#  endif
#else
#  define USE_TDM 0
#endif

union FragBF {
    bf16x16 v;
    uint4   u[2];   // u[0] = elements 0..7, u[1] = elements 8..15
};

__device__ __forceinline__ unsigned short f2bf(float f) {
    __bf16 h = (__bf16)f;                       // native v_cvt, RNE
    return __builtin_bit_cast(unsigned short, h);
}
__device__ __forceinline__ float bf2f(unsigned short h) {
    return (float)__builtin_bit_cast(__bf16, h);
}

// xor-butterfly within each 16-lane half via v_permlane16_b32 (VALU, no LDS).
// select nibble for lane i is (i ^ mask).
__device__ __forceinline__ float permx(float v, unsigned lo, unsigned hi) {
    int s = __builtin_bit_cast(int, v);
    int r = __builtin_amdgcn_permlane16(s, s, (int)lo, (int)hi, false, false);
    return __builtin_bit_cast(float, r);
}
__device__ __forceinline__ float rowmax16(float v) {
    v = fmaxf(v, permx(v, 0x67452301u, 0xEFCDAB89u)); // xor 1
    v = fmaxf(v, permx(v, 0x54761032u, 0xDCFE98BAu)); // xor 2
    v = fmaxf(v, permx(v, 0x32107654u, 0xBA98FEDCu)); // xor 4
    v = fmaxf(v, permx(v, 0xFEDCBA98u, 0x76543210u)); // xor 8
    return v;
}
__device__ __forceinline__ float rowsum16(float v) {
    v += permx(v, 0x67452301u, 0xEFCDAB89u);
    v += permx(v, 0x54761032u, 0xDCFE98BAu);
    v += permx(v, 0x32107654u, 0xBA98FEDCu);
    v += permx(v, 0xFEDCBA98u, 0x76543210u);
    return v;
}

#if USE_TDM
// Issue one TDM 2D tile load: rows x cols of bf16, global row stride = stride0
// elements, into LDS at ldsAddr with +4 DWORDs padding after every 32 DWORDs
// (tile row = 64 bf16 = 32 DWORDs -> LDS row stride 72 bf16 = LDP).
// D# layout per CDNA5 ISA.
__device__ __forceinline__ void tdm_load_2d(const void* gsrc, unsigned ldsAddr,
                                            unsigned dim0, unsigned dim1,
                                            unsigned stride0,
                                            unsigned tile0, unsigned tile1) {
    unsigned long long ga = (unsigned long long)(size_t)gsrc;
    uint4v g0;
    g0[0] = 1u;                                   // count=1 (valid), user mode
    g0[1] = ldsAddr;                              // LDS byte address
    g0[2] = (unsigned)ga;                         // global addr [95:64]
    g0[3] = (unsigned)((ga >> 32) & 0x01FFFFFFull) | (2u << 30); // addr hi, type=2
    int8v g1;
    g1[0] = (int)((1u << 16)      // data_size = 2 bytes
                | (1u << 20)      // pad_enable
                | (4u << 22)      // pad_interval: 32 DWORDs
                | (3u << 25));    // pad_amount: 4 DWORDs
    g1[1] = (int)((dim0 & 0xFFFFu) << 16);                       // tensor_dim0 lo
    g1[2] = (int)((dim0 >> 16) | ((dim1 & 0xFFFFu) << 16));      // dim0 hi, dim1 lo
    g1[3] = (int)((dim1 >> 16) | (tile0 << 16));                 // dim1 hi, tile_dim0
    g1[4] = (int)tile1;                                          // tile_dim1 (tile_dim2=0)
    g1[5] = (int)stride0;                                        // tensor_dim0_stride lo
    g1[6] = 0;
    g1[7] = 0;
    int4v z4 = {0, 0, 0, 0};
#if __clang_major__ >= 23
    int8v z8 = {0, 0, 0, 0, 0, 0, 0, 0};
    __builtin_amdgcn_tensor_load_to_lds(g0, g1, z4, z4, z8, 0);
#else
    __builtin_amdgcn_tensor_load_to_lds(g0, g1, z4, z4, 0);
#endif
}
#endif

// ------------------------- fp32 -> bf16 weight convert ----------------------
__global__ void cvt_bf16_kernel(const float4* __restrict__ in,
                                ushort4* __restrict__ out, int n4) {
    int i = blockIdx.x * 256 + threadIdx.x;
    if (i < n4) {
        float4 v = in[i];
        ushort4 r;
        r.x = f2bf(v.x); r.y = f2bf(v.y); r.z = f2bf(v.z); r.w = f2bf(v.w);
        out[i] = r;
    }
}

// ------------------------------ LayerNorm ----------------------------------
__global__ __launch_bounds__(256) void ln_kernel(
    const float* __restrict__ x, const float* __restrict__ g,
    const float* __restrict__ b, unsigned short* __restrict__ out) {
    __shared__ float s1[256], s2[256];
    const int row = blockIdx.x;
    const float* xr = x + (size_t)row * 1024;
    float4 v = ((const float4*)xr)[threadIdx.x];
    s1[threadIdx.x] = v.x + v.y + v.z + v.w;
    s2[threadIdx.x] = v.x*v.x + v.y*v.y + v.z*v.z + v.w*v.w;
    __syncthreads();
    for (int st = 128; st > 0; st >>= 1) {
        if (threadIdx.x < st) {
            s1[threadIdx.x] += s1[threadIdx.x + st];
            s2[threadIdx.x] += s2[threadIdx.x + st];
        }
        __syncthreads();
    }
    float mu   = s1[0] * (1.0f / 1024.0f);
    float var  = s2[0] * (1.0f / 1024.0f) - mu * mu;
    float rstd = rsqrtf(var + 1e-6f);
    float4 gg = ((const float4*)g)[threadIdx.x];
    float4 bb = ((const float4*)b)[threadIdx.x];
    ushort4 r;
    r.x = f2bf((v.x - mu) * rstd * gg.x + bb.x);
    r.y = f2bf((v.y - mu) * rstd * gg.y + bb.y);
    r.z = f2bf((v.z - mu) * rstd * gg.z + bb.z);
    r.w = f2bf((v.w - mu) * rstd * gg.w + bb.w);
    *(ushort4*)(out + (size_t)row * 1024 + threadIdx.x * 4) = r;
}

// ------------------------------- GEMM --------------------------------------
// out[m,n] = sum_k A[m,k] * W[n,k] (+ bias / residual / GELU epilogue).
// Block tile 128x128, BK=64, 8 waves of 64x32, v_wmma_f32_16x16x32_bf16.
// TDM double buffering: wave 0 issues tensor_load_to_lds for tile kt+1 into
// the alternate buffer before the 16-WMMA compute block, then waits TENSORcnt
// and the block barriers.  (Fallback: register-staged copies.)
#define BM  128
#define BN  128
#define BK  64
#define LDP 72   // padded LDS row stride (bf16 units) -> 144B, 16B aligned

enum { EPI_BF16 = 0, EPI_F32_RES = 1, EPI_BF16_GELU = 2 };

template<int EPI>
__global__ __launch_bounds__(256) void gemm_bf16_kernel(
    const unsigned short* __restrict__ A,   // [M,K] bf16
    const unsigned short* __restrict__ W,   // [N,K] bf16
    const float* __restrict__ bias,         // [N]
    const float* __restrict__ res,          // [M,N] (EPI_F32_RES only)
    float* __restrict__ outF,               // [M,N] (EPI_F32_RES only)
    unsigned short* __restrict__ outB,      // [M,N] bf16 (other epilogues)
    int M, int N, int K)
{
    __shared__ unsigned short As[2][BM * LDP];
    __shared__ unsigned short Bs[2][BN * LDP];
    const int tid  = threadIdx.x;
    const int lane = tid & 31;
    const int wave = tid >> 5;
    const int r16  = lane & 15;
    const int hi   = lane >> 4;
    const int m0   = blockIdx.y * BM;
    const int n0   = blockIdx.x * BN;
    const int wm0  = (wave >> 2) * 64;   // wave M origin within block tile
    const int wn0  = (wave & 3)  * 32;   // wave N origin within block tile

    fx8 acc[4][2];
    #pragma unroll
    for (int i = 0; i < 4; i++)
        #pragma unroll
        for (int j = 0; j < 2; j++)
            #pragma unroll
            for (int e = 0; e < 8; e++) acc[i][j][e] = 0.0f;

    const int kTiles = K / BK;

#if USE_TDM
    auto issueTiles = [&](int kt, int buf) {
        const int k0 = kt * BK;
        tdm_load_2d(&A[(size_t)m0 * K + k0],
                    (unsigned)(size_t)(void*)&As[buf][0],
                    (unsigned)(K - k0), (unsigned)(M - m0), (unsigned)K, BK, BM);
        tdm_load_2d(&W[(size_t)n0 * K + k0],
                    (unsigned)(size_t)(void*)&Bs[buf][0],
                    (unsigned)(K - k0), (unsigned)(N - n0), (unsigned)K, BK, BN);
    };
    if (wave == 0) {
        issueTiles(0, 0);
        __builtin_amdgcn_s_wait_tensorcnt(0);
    }
    __syncthreads();
#else
    const int rr0 = tid >> 3;            // staging row 0..31 (+32*c)
    const int kc8 = (tid & 7) * 8;       // staging K offset (bf16 units)
    uint4 ra[4], rb[4];
    auto loadTile = [&](int kt) {
        const int k0 = kt * BK;
        #pragma unroll
        for (int c = 0; c < 4; c++) {
            ra[c] = *(const uint4*)&A[(size_t)(m0 + rr0 + 32*c) * K + k0 + kc8];
            rb[c] = *(const uint4*)&W[(size_t)(n0 + rr0 + 32*c) * K + k0 + kc8];
        }
    };
    auto storeTile = [&](int buf) {
        #pragma unroll
        for (int c = 0; c < 4; c++) {
            *(uint4*)&As[buf][(rr0 + 32*c) * LDP + kc8] = ra[c];
            *(uint4*)&Bs[buf][(rr0 + 32*c) * LDP + kc8] = rb[c];
        }
    };
    loadTile(0);
    storeTile(0);
    __syncthreads();
#endif

    for (int kt = 0; kt < kTiles; ++kt) {
        const int cur = kt & 1;
        const bool hasNext = (kt + 1 < kTiles);
#if USE_TDM
        // Prior readers of buffer cur^1 synced at end of iteration kt-1, so
        // the DMA into it can run concurrently with this tile's WMMAs.
        if (hasNext && wave == 0) issueTiles(kt + 1, cur ^ 1);
#else
        if (hasNext) loadTile(kt + 1);    // global loads overlap WMMA below
#endif
        #pragma unroll
        for (int ks = 0; ks < 2; ks++) {  // two K=32 WMMA steps
            FragBF aF[4], bF[2];
            #pragma unroll
            for (int i = 0; i < 4; i++) { // A frag: lane=row, interleaved K
                const unsigned short* p =
                    &As[cur][(wm0 + i * 16 + r16) * LDP + ks * 32 + hi * 8];
                aF[i].u[0] = *(const uint4*)p;         // K = (hi?8:0)+0..7
                aF[i].u[1] = *(const uint4*)(p + 16);  // K = 16+(hi?8:0)+0..7
            }
            #pragma unroll
            for (int j = 0; j < 2; j++) { // B frag: lane=col, interleaved K
                const unsigned short* p =
                    &Bs[cur][(wn0 + j * 16 + r16) * LDP + ks * 32 + hi * 8];
                bF[j].u[0] = *(const uint4*)p;         // K = (hi?8:0)+0..7
                bF[j].u[1] = *(const uint4*)(p + 16);  // K = 16+(hi?8:0)+0..7
            }
            #pragma unroll
            for (int i = 0; i < 4; i++)
                #pragma unroll
                for (int j = 0; j < 2; j++)
                    acc[i][j] = __builtin_amdgcn_wmma_f32_16x16x32_bf16(
                        false, aF[i].v, false, bF[j].v,
                        (short)0, acc[i][j], false, false);
        }
#if USE_TDM
        if (hasNext && wave == 0) __builtin_amdgcn_s_wait_tensorcnt(0);
#else
        if (hasNext) storeTile(cur ^ 1);  // prior readers of buf cur^1 synced
#endif
        __syncthreads();
    }

    // Epilogue. C/D layout: lane (c=l&15, hi), VGPR v -> (row = v+8*hi, col=c).
    #pragma unroll
    for (int i = 0; i < 4; i++) {
        #pragma unroll
        for (int j = 0; j < 2; j++) {
            const int col = n0 + wn0 + j * 16 + r16;
            const float bv = bias[col];
            #pragma unroll
            for (int v = 0; v < 8; v++) {
                const int row = m0 + wm0 + i * 16 + v + hi * 8;
                const size_t idx = (size_t)row * N + col;
                float val = acc[i][j][v] + bv;
                if constexpr (EPI == EPI_F32_RES) {
                    outF[idx] = val + res[idx];
                } else if constexpr (EPI == EPI_BF16_GELU) {
                    float g = 0.5f * val * (1.0f + erff(val * 0.70710678118654752f));
                    outB[idx] = f2bf(g);
                } else {
                    outB[idx] = f2bf(val);
                }
            }
        }
    }
}

// ----------------------- QKV split / head reshape ---------------------------
// qkv [B*N, 3C] bf16 -> q,k [BH, N, HD] bf16 (q pre-scaled by HD^-0.5),
// vT [BH, HD, N] bf16 (transposed so PV B-frags are contiguous).
__global__ void qkv_split_kernel(const unsigned short* __restrict__ qkv,
                                 unsigned short* __restrict__ q,
                                 unsigned short* __restrict__ k,
                                 unsigned short* __restrict__ vT) {
    int id = blockIdx.x * 256 + threadIdx.x;   // over B*N*C = 4M
    int t = id >> 10;        // token 0..4095
    int c = id & 1023;
    int b = t >> 10;
    int n = t & 1023;
    int h = c >> 6;
    int d = c & 63;
    const unsigned short* row = qkv + (size_t)t * 3072;
    size_t bh = (size_t)(b * 16 + h);
    q[(bh * 1024 + n) * 64 + d] = f2bf(bf2f(row[c]) * 0.125f);  // 1/sqrt(64)
    k[(bh * 1024 + n) * 64 + d] = row[1024 + c];
    vT[(bh * 64 + d) * 1024 + n] = row[2048 + c];
}

// --------------------------- Flash attention --------------------------------
// One wave owns 16 query rows of one (b,h). Key blocks of 32; QK^T = 2 WMMAs
// per 16-key tile; online softmax with permlane16 butterflies (row r = j+8*hi
// lives in 16 lanes of one half-wave, so stats stay lane-local); P re-laid-out
// through LDS into an A-frag; PV = 4 WMMAs per key block.
__global__ __launch_bounds__(256) void attn_kernel(
    const unsigned short* __restrict__ q,    // [BH, N, HD]
    const unsigned short* __restrict__ k,    // [BH, N, HD]
    const unsigned short* __restrict__ vT,   // [BH, HD, N]
    unsigned short* __restrict__ o)          // [B*N, C] bf16
{
    __shared__ unsigned short pLds[8 * 16 * 32];
    const int lane = threadIdx.x & 31;
    const int wave = threadIdx.x >> 5;
    const int r16  = lane & 15;
    const int hi   = lane >> 4;
    const int wid  = blockIdx.x * 8 + wave;  // 0..4095
    const int qt   = wid & 63;               // query tile within sequence
    const int bh   = wid >> 6;               // 0..63
    const int h    = bh & 15;
    const int b    = bh >> 4;

    FragBF qF[2];
    const unsigned short* qbase = q + ((size_t)bh * 1024 + qt * 16 + r16) * 64;
    #pragma unroll
    for (int ks = 0; ks < 2; ks++) {
        qF[ks].u[0] = *(const uint4*)(qbase + ks * 32 + hi * 8);
        qF[ks].u[1] = *(const uint4*)(qbase + ks * 32 + 16 + hi * 8);
    }

    fx8 acc[4];
    #pragma unroll
    for (int t = 0; t < 4; t++)
        #pragma unroll
        for (int e = 0; e < 8; e++) acc[t][e] = 0.0f;
    float mrun[8], lrun[8];
    #pragma unroll
    for (int j = 0; j < 8; j++) { mrun[j] = -3.0e38f; lrun[j] = 0.0f; }

    unsigned short* pw = &pLds[wave * 16 * 32];

    for (int kb = 0; kb < 32; ++kb) {            // 32 key blocks of 32
        fx8 S[2];
        #pragma unroll
        for (int t = 0; t < 2; t++) {            // two 16-key score tiles
            FragBF kF0, kF1;
            const unsigned short* kp =
                k + ((size_t)bh * 1024 + kb * 32 + t * 16 + r16) * 64;
            kF0.u[0] = *(const uint4*)(kp + hi * 8);
            kF0.u[1] = *(const uint4*)(kp + hi * 8 + 16);
            kF1.u[0] = *(const uint4*)(kp + 32 + hi * 8);
            kF1.u[1] = *(const uint4*)(kp + 32 + hi * 8 + 16);
            fx8 s;
            #pragma unroll
            for (int e = 0; e < 8; e++) s[e] = 0.0f;
            s = __builtin_amdgcn_wmma_f32_16x16x32_bf16(
                false, qF[0].v, false, kF0.v, (short)0, s, false, false);
            s = __builtin_amdgcn_wmma_f32_16x16x32_bf16(
                false, qF[1].v, false, kF1.v, (short)0, s, false, false);
            S[t] = s;
        }
        // Online softmax with pure-VALU butterflies.
        float al[8];
        #pragma unroll
        for (int j = 0; j < 8; j++) {
            float rm    = rowmax16(fmaxf(S[0][j], S[1][j]));
            float mnew  = fmaxf(mrun[j], rm);
            float alpha = __expf(mrun[j] - mnew);
            float p0 = __expf(S[0][j] - mnew);
            float p1 = __expf(S[1][j] - mnew);
            float rs = rowsum16(p0 + p1);
            lrun[j] = lrun[j] * alpha + rs;
            mrun[j] = mnew;
            al[j]   = alpha;
            pw[(j + 8 * hi) * 32 + r16]      = f2bf(p0);  // P cols 0..15
            pw[(j + 8 * hi) * 32 + 16 + r16] = f2bf(p1);  // P cols 16..31
        }
        #pragma unroll
        for (int t = 0; t < 4; t++)
            #pragma unroll
            for (int j = 0; j < 8; j++) acc[t][j] *= al[j];
        asm volatile("s_wait_dscnt 0" ::: "memory");
        FragBF pF;   // P (16x32) as A-frag from LDS
        pF.u[0] = *(const uint4*)&pw[r16 * 32 + hi * 8];
        pF.u[1] = *(const uint4*)&pw[r16 * 32 + 16 + hi * 8];
        #pragma unroll
        for (int t = 0; t < 4; t++) {            // PV over HD in 16-col tiles
            FragBF vF;
            const unsigned short* vp =
                vT + ((size_t)bh * 64 + t * 16 + r16) * 1024 + kb * 32 + hi * 8;
            vF.u[0] = *(const uint4*)vp;
            vF.u[1] = *(const uint4*)(vp + 16);
            acc[t] = __builtin_amdgcn_wmma_f32_16x16x32_bf16(
                false, pF.v, false, vF.v, (short)0, acc[t], false, false);
        }
    }
    float inv[8];
    #pragma unroll
    for (int j = 0; j < 8; j++) inv[j] = 1.0f / lrun[j];
    #pragma unroll
    for (int t = 0; t < 4; t++)
        #pragma unroll
        for (int j = 0; j < 8; j++) {
            int row = qt * 16 + j + 8 * hi;
            o[((size_t)(b * 1024) + row) * 1024 + h * 64 + t * 16 + r16] =
                f2bf(acc[t][j] * inv[j]);
        }
}

// --------------------------------------------------------------------------
extern "C" void kernel_launch(void* const* d_in, const int* in_sizes, int n_in,
                              void* d_out, int out_size, void* d_ws, size_t ws_size,
                              hipStream_t stream) {
    const float* x       = (const float*)d_in[0];
    const float* qkv_w   = (const float*)d_in[1];
    const float* qkv_b   = (const float*)d_in[2];
    const float* ap_w    = (const float*)d_in[3];
    const float* ap_b    = (const float*)d_in[4];
    const float* bp_w    = (const float*)d_in[5];
    const float* bp_b    = (const float*)d_in[6];
    const float* ln1_g   = (const float*)d_in[7];
    const float* ln1_b   = (const float*)d_in[8];
    const float* ln2_g   = (const float*)d_in[9];
    const float* ln2_b   = (const float*)d_in[10];
    const float* fc1_w   = (const float*)d_in[11];
    const float* fc1_b   = (const float*)d_in[12];
    const float* fc2_w   = (const float*)d_in[13];
    const float* fc2_b   = (const float*)d_in[14];

    char* wp = (char*)d_ws;
    auto alloc = [&](size_t bytes) -> char* {
        char* p = wp; wp += (bytes + 255) & ~(size_t)255; return p;
    };
    const size_t M = 4096;  // B*N
    unsigned short* qkvw_bf = (unsigned short*)alloc(3145728ull * 2);
    unsigned short* apw_bf  = (unsigned short*)alloc(1048576ull * 2);
    unsigned short* bpw_bf  = (unsigned short*)alloc(1048576ull * 2);
    unsigned short* fc1w_bf = (unsigned short*)alloc(4194304ull * 2);
    unsigned short* fc2w_bf = (unsigned short*)alloc(4194304ull * 2);
    unsigned short* xn1     = (unsigned short*)alloc(M * 1024 * 2);
    unsigned short* qkvbuf  = (unsigned short*)alloc(M * 3072 * 2);
    unsigned short* qb      = (unsigned short*)alloc(M * 1024 * 2);
    unsigned short* kb      = (unsigned short*)alloc(M * 1024 * 2);
    unsigned short* vTb     = (unsigned short*)alloc(M * 1024 * 2);
    unsigned short* ob      = (unsigned short*)alloc(M * 1024 * 2);
    unsigned short* t1      = (unsigned short*)alloc(M * 1024 * 2);
    float*          x2      = (float*)alloc(M * 1024 * 4);
    unsigned short* xn2     = (unsigned short*)alloc(M * 1024 * 2);
    unsigned short* h1      = (unsigned short*)alloc(M * 4096 * 2);

    // 1) weights -> bf16 (vectorized: n/4 float4s)
    cvt_bf16_kernel<<<3072, 256, 0, stream>>>((const float4*)qkv_w, (ushort4*)qkvw_bf, 786432);
    cvt_bf16_kernel<<<1024, 256, 0, stream>>>((const float4*)ap_w,  (ushort4*)apw_bf,  262144);
    cvt_bf16_kernel<<<1024, 256, 0, stream>>>((const float4*)bp_w,  (ushort4*)bpw_bf,  262144);
    cvt_bf16_kernel<<<4096, 256, 0, stream>>>((const float4*)fc1_w, (ushort4*)fc1w_bf, 1048576);
    cvt_bf16_kernel<<<4096, 256, 0, stream>>>((const float4*)fc2_w, (ushort4*)fc2w_bf, 1048576);

    // 2) LN1
    ln_kernel<<<4096, 256, 0, stream>>>(x, ln1_g, ln1_b, xn1);

    // 3) QKV GEMM: [4096,1024] x [3072,1024]^T
    gemm_bf16_kernel<EPI_BF16><<<dim3(24, 32), 256, 0, stream>>>(
        xn1, qkvw_bf, qkv_b, nullptr, nullptr, qkvbuf, 4096, 3072, 1024);

    // 4) split heads (+ q scale, v transpose)
    qkv_split_kernel<<<16384, 256, 0, stream>>>(qkvbuf, qb, kb, vTb);

    // 5) flash attention -> o [B*N, C]
    attn_kernel<<<512, 256, 0, stream>>>(qb, kb, vTb, ob);

    // 6) attention-internal projection
    gemm_bf16_kernel<EPI_BF16><<<dim3(8, 32), 256, 0, stream>>>(
        ob, apw_bf, ap_b, nullptr, nullptr, t1, 4096, 1024, 1024);

    // 7) block projection + residual -> x2 (fp32)
    gemm_bf16_kernel<EPI_F32_RES><<<dim3(8, 32), 256, 0, stream>>>(
        t1, bpw_bf, bp_b, x, x2, nullptr, 4096, 1024, 1024);

    // 8) LN2
    ln_kernel<<<4096, 256, 0, stream>>>(x2, ln2_g, ln2_b, xn2);

    // 9) FC1 + exact GELU
    gemm_bf16_kernel<EPI_BF16_GELU><<<dim3(32, 32), 256, 0, stream>>>(
        xn2, fc1w_bf, fc1_b, nullptr, nullptr, h1, 4096, 4096, 1024);

    // 10) FC2 + residual -> output (fp32)
    gemm_bf16_kernel<EPI_F32_RES><<<dim3(8, 32), 256, 0, stream>>>(
        h1, fc2w_bf, fc2_b, x2, (float*)d_out, nullptr, 4096, 1024, 4096);
}
